// GraphConvModel_47459388620990
// MI455X (gfx1250) — matrix operand order repaired and checked by reading, not compiled, not run on hardware
//
#include <hip/hip_runtime.h>
#include <hip/hip_bf16.h>
#include <math.h>

typedef __attribute__((ext_vector_type(16))) __bf16 v16bf;
typedef __attribute__((ext_vector_type(8)))  __bf16 v8bf;
typedef __attribute__((ext_vector_type(8)))  float  v8f;

#define N_ATOMS 1000000
#define NBLK    62500          // N_ATOMS / 16
#define BATCH   40000
#define EPSBN   1e-3f

union FragBF { v16bf v; v8bf h[2]; };

__device__ __forceinline__ __bf16 to_bf16(float f) {
  unsigned u = __float_as_uint(f);
  u += 0x7FFFu + ((u >> 16) & 1u);          // round-to-nearest-even
  unsigned short h = (unsigned short)(u >> 16);
  __bf16 b; __builtin_memcpy(&b, &h, 2); return b;
}

// Branchless tanh: sign(x) * (1 - 2/(exp(2|x|)+1)).
// v_exp_f32 + v_rcp_f32, no EXEC divergence; exp(inf)->inf -> rcp->0 -> +/-1.
__device__ __forceinline__ float fast_tanh(float x) {
  float ax = __builtin_fabsf(x);
  float e  = __expf(2.0f * ax);
  float r  = 1.0f - 2.0f * __builtin_amdgcn_rcpf(e + 1.0f);
  return __builtin_copysignf(r, x);
}

// Order-preserving float <-> uint mapping: float compare == unsigned compare.
__device__ __forceinline__ unsigned fkey(float v) {
  unsigned u = __float_as_uint(v);
  unsigned mask = (unsigned)((int)u >> 31);          // all-ones if negative
  return u ^ (mask | 0x80000000u);
}
__device__ __forceinline__ float funkey(unsigned key) {
  unsigned mask = (unsigned)(~((int)key >> 31)) | 0x80000000u;
  return __uint_as_float(key ^ mask);
}
#define NEG_INF_KEY 0x007FFFFFu                      // fkey(-inf)

// degree of a 16-row block (all boundaries are multiples of 16)
__device__ __forceinline__ void block_degree(int row, int& d, int& start) {
  if (row < 10000)       { d = 0; start = 0;      }
  else if (row < 250000) { d = 1; start = 10000;  }
  else if (row < 610000) { d = 2; start = 250000; }
  else if (row < 860000) { d = 3; start = 610000; }
  else                   { d = 4; start = 860000; }
}

// ---------------------------------------------------------------------------
// Pre-swizzle f32 weights [nmats][IN][OUT] into WMMA B-fragment lane order.
// Fragment chunk (mat,kt,nt): 32 lanes x 16 bf16; lane L holds col nt*16+(L&15),
// k = kt*32 + ((L>=16)?8:0) + (e&7) + ((e>=8)?16:0)  for e = 0..15 (table order).
// ---------------------------------------------------------------------------
__global__ void prep_frags(const float* __restrict__ W, __bf16* __restrict__ out,
                           int IN, int OUT, int KT, int nmats) {
  int NT = OUT / 16;
  long total = (long)nmats * KT * NT * 512;
  for (long t = (long)blockIdx.x * 256 + threadIdx.x; t < total;
       t += (long)gridDim.x * 256) {
    int e = (int)(t & 15);
    long r = t >> 4;
    int lane = (int)(r & 31); r >>= 5;
    int nt = (int)(r % NT);   r /= NT;
    int kt = (int)(r % KT);
    int mat = (int)(r / KT);
    int kbase = (lane >= 16) ? 8 : 0;
    int k = kt * 32 + kbase + (e & 7) + ((e & 8) ? 16 : 0);
    int n = nt * 16 + (lane & 15);
    float v = (k < IN) ? W[(long)mat * IN * OUT + (long)k * OUT + n] : 0.0f;
    out[t] = to_bf16(v);
  }
}

// ---------------------------------------------------------------------------
// Fused graph-conv layer: per 16-row degree-uniform tile:
//   stage self rows + neighbor-sum rows in LDS (bf16),
//   C = Aself*Wself[d] + Anbr*Wrel[d-1] + b[d]  via v_wmma_f32_16x16x32_bf16,
//   out = bn1(tanh(C))
// ---------------------------------------------------------------------------
template<int IN, int KPAD>
__global__ void __launch_bounds__(256)
gc_kernel(const float* __restrict__ x,
          const int* __restrict__ adj1, const int* __restrict__ adj2,
          const int* __restrict__ adj3, const int* __restrict__ adj4,
          const __bf16* __restrict__ frS, const __bf16* __restrict__ frR,
          const float* __restrict__ bias,
          const float* __restrict__ gamma, const float* __restrict__ beta,
          const float* __restrict__ mean,  const float* __restrict__ var,
          float* __restrict__ out)
{
  constexpr int OUT = 128, NT = 8, KT = KPAD / 32;
  const int tid = threadIdx.x;
  const int rowBase = blockIdx.x * 16;
  int d, start;
  block_degree(rowBase, d, start);
  const int* __restrict__ adj = (d == 1) ? adj1 : (d == 2) ? adj2
                              : (d == 3) ? adj3 : adj4;

  __shared__ __align__(16) __bf16 As[16][KPAD];
  __shared__ __align__(16) __bf16 An[16][KPAD];
  __shared__ int nidx[64];

  if (d > 0 && tid < 16 * d) {
    int r = tid / d, j = tid % d;
    nidx[r * d + j] = adj[(long)(rowBase - start + r) * d + j];
  }
  __syncthreads();

  for (int t = tid; t < 16 * KPAD; t += 256) {
    int r = t / KPAD, c = t % KPAD;
    float sv = 0.f, nv = 0.f;
    if (c < IN) {
      sv = x[(long)(rowBase + r) * IN + c];
      for (int j = 0; j < d; ++j)
        nv += x[(long)nidx[r * d + j] * IN + c];
    }
    As[r][c] = to_bf16(sv);
    An[r][c] = to_bf16(nv);
  }
  __syncthreads();

  const int wave = tid >> 5, lane = tid & 31;
  const int col  = wave * 16 + (lane & 15);
  const int m    = lane & 15;
  const int abk  = (lane & 16) ? 8 : 0;

  float bv = bias[d * OUT + col];
  v8f c = { bv, bv, bv, bv, bv, bv, bv, bv };

  const __bf16* fs = frS + ((long)d * KT * NT + wave) * 512 + lane * 16;
  const __bf16* fn = (d > 0)
      ? frR + ((long)(d - 1) * KT * NT + wave) * 512 + lane * 16 : frR;

  for (int kt = 0; kt < KT; ++kt) {
    FragBF a, b;
    a.h[0] = *(const v8bf*)&As[m][kt * 32 + abk];
    a.h[1] = *(const v8bf*)&As[m][kt * 32 + abk + 16];
    const __bf16* bp = fs + (long)kt * NT * 512;
    b.h[0] = *(const v8bf*)(bp);
    b.h[1] = *(const v8bf*)(bp + 8);
    c = __builtin_amdgcn_wmma_f32_16x16x32_bf16(false, a.v, false, b.v,
                                                (short)0, c, false, false);
    if (d > 0) {
      FragBF an, bn;
      an.h[0] = *(const v8bf*)&An[m][kt * 32 + abk];
      an.h[1] = *(const v8bf*)&An[m][kt * 32 + abk + 16];
      const __bf16* bq = fn + (long)kt * NT * 512;
      bn.h[0] = *(const v8bf*)(bq);
      bn.h[1] = *(const v8bf*)(bq + 8);
      c = __builtin_amdgcn_wmma_f32_16x16x32_bf16(false, an.v, false, bn.v,
                                                  (short)0, c, false, false);
    }
  }

  float g  = gamma[col], bt = beta[col], mu = mean[col];
  float rs = __builtin_amdgcn_rsqf(var[col] + EPSBN);
  const int mbase = rowBase + ((lane & 16) ? 8 : 0);
#pragma unroll
  for (int i = 0; i < 8; ++i) {
    float v = fast_tanh(c[i]);
    v = (v - mu) * rs * g + bt;
    out[(long)(mbase + i) * OUT + col] = v;
  }
}

// ---------------------------------------------------------------------------
// Graph pool: out[i] = max(x[i], x[neighbors(i)...]) per 128-wide row, float4.
// ---------------------------------------------------------------------------
__global__ void __launch_bounds__(256)
pool_kernel(const float* __restrict__ x,
            const int* __restrict__ adj1, const int* __restrict__ adj2,
            const int* __restrict__ adj3, const int* __restrict__ adj4,
            float* __restrict__ out)
{
  long g = (long)blockIdx.x * 256 + threadIdx.x;
  if (g >= (long)N_ATOMS * 32) return;
  int atom = (int)(g >> 5);
  int ch   = (int)(g & 31);
  int d, start; block_degree(atom, d, start);
  float4 v = ((const float4*)(x + (long)atom * 128))[ch];
  if (d > 0) {
    const int* adj = (d == 1) ? adj1 : (d == 2) ? adj2 : (d == 3) ? adj3 : adj4;
    int r = atom - start;
    for (int j = 0; j < d; ++j) {
      int nb = adj[(long)r * d + j];
      float4 w = ((const float4*)(x + (long)nb * 128))[ch];
      v.x = fmaxf(v.x, w.x); v.y = fmaxf(v.y, w.y);
      v.z = fmaxf(v.z, w.z); v.w = fmaxf(v.w, w.w);
    }
  }
  ((float4*)(out + (long)atom * 128))[ch] = v;
}

// ---------------------------------------------------------------------------
// Fused dense(128->256) + tanh + bn3 + segment sum/max via L2-resident atomics.
// Max uses order-preserving uint keys -> single branchless atomicMax(u32).
// ---------------------------------------------------------------------------
__global__ void __launch_bounds__(256)
dense_gather_kernel(const float* __restrict__ x,
                    const int* __restrict__ membership,
                    const __bf16* __restrict__ frW,
                    const float* __restrict__ bias,
                    const float* __restrict__ gamma, const float* __restrict__ beta,
                    const float* __restrict__ mean,  const float* __restrict__ var,
                    float* __restrict__ acc_sum, unsigned* __restrict__ acc_max)
{
  constexpr int IN = 128, NT = 16, KT = 4;
  const int tid = threadIdx.x;
  const int rowBase = blockIdx.x * 16;

  __shared__ __align__(16) __bf16 As[16][IN];
  __shared__ int memb[16];

  if (tid < 16) memb[tid] = membership[rowBase + tid];
  for (int t = tid; t < 16 * IN; t += 256) {
    int r = t / IN, cc = t % IN;
    As[r][cc] = to_bf16(x[(long)(rowBase + r) * IN + cc]);
  }
  __syncthreads();

  const int wave = tid >> 5, lane = tid & 31;
  const int m    = lane & 15;
  const int abk  = (lane & 16) ? 8 : 0;
  const int moff = (lane & 16) ? 8 : 0;

  for (int half = 0; half < 2; ++half) {
    int nt  = wave + half * 8;
    int col = nt * 16 + (lane & 15);
    float bv = bias[col];
    v8f c = { bv, bv, bv, bv, bv, bv, bv, bv };
    for (int kt = 0; kt < KT; ++kt) {
      FragBF a, b;
      a.h[0] = *(const v8bf*)&As[m][kt * 32 + abk];
      a.h[1] = *(const v8bf*)&As[m][kt * 32 + abk + 16];
      const __bf16* bp = frW + ((long)kt * NT + nt) * 512 + lane * 16;
      b.h[0] = *(const v8bf*)(bp);
      b.h[1] = *(const v8bf*)(bp + 8);
      c = __builtin_amdgcn_wmma_f32_16x16x32_bf16(false, a.v, false, b.v,
                                                  (short)0, c, false, false);
    }
    float gmm = gamma[col], bt = beta[col], mu = mean[col];
    float rs = __builtin_amdgcn_rsqf(var[col] + EPSBN);
#pragma unroll
    for (int i = 0; i < 8; ++i) {
      float v = fast_tanh(c[i]);
      v = (v - mu) * rs * gmm + bt;
      int seg = memb[moff + i];
      atomicAdd(&acc_sum[(long)seg * 256 + col], v);
      atomicMax(&acc_max[(long)seg * 256 + col], fkey(v));
    }
  }
}

__global__ void init_accum(float* __restrict__ acc_sum, unsigned* __restrict__ acc_max) {
  long i = (long)blockIdx.x * 256 + threadIdx.x;
  if (i < (long)BATCH * 256) { acc_sum[i] = 0.0f; acc_max[i] = NEG_INF_KEY; }
}

__global__ void finalize_kernel(const float* __restrict__ acc_sum,
                                const unsigned* __restrict__ acc_max,
                                float* __restrict__ out) {
  long i = (long)blockIdx.x * 256 + threadIdx.x;
  if (i >= (long)BATCH * 512) return;
  int g  = (int)(i >> 9);
  int cc = (int)(i & 511);
  float v = (cc < 256) ? acc_sum[(long)g * 256 + cc]
                       : funkey(acc_max[(long)g * 256 + (cc - 256)]);
  out[i] = fast_tanh(v);
}

// ---------------------------------------------------------------------------
extern "C" void kernel_launch(void* const* d_in, const int* in_sizes, int n_in,
                              void* d_out, int out_size, void* d_ws, size_t ws_size,
                              hipStream_t stream)
{
  (void)in_sizes; (void)n_in; (void)out_size; (void)ws_size;
  const float* atom_features = (const float*)d_in[0];
  const int*   membership    = (const int*)d_in[1];
  const int*   adj1 = (const int*)d_in[2];
  const int*   adj2 = (const int*)d_in[3];
  const int*   adj3 = (const int*)d_in[4];
  const int*   adj4 = (const int*)d_in[5];
  const float* gc1_Wself = (const float*)d_in[6];
  const float* gc1_Wrel  = (const float*)d_in[7];
  const float* gc1_b     = (const float*)d_in[8];
  const float* gc2_Wself = (const float*)d_in[9];
  const float* gc2_Wrel  = (const float*)d_in[10];
  const float* gc2_b     = (const float*)d_in[11];
  const float* bn1_gamma = (const float*)d_in[12];
  const float* bn1_beta  = (const float*)d_in[13];
  const float* bn1_mean  = (const float*)d_in[14];
  const float* bn1_var   = (const float*)d_in[15];
  const float* bn3_gamma = (const float*)d_in[16];
  const float* bn3_beta  = (const float*)d_in[17];
  const float* bn3_mean  = (const float*)d_in[18];
  const float* bn3_var   = (const float*)d_in[19];
  const float* dense_W   = (const float*)d_in[20];
  const float* dense_b   = (const float*)d_in[21];

  char* ws = (char*)d_ws;
  float*    bufA    = (float*)(ws);                     // 1M x 128 f32 (512 MB)
  float*    bufB    = (float*)(ws + 512000000ULL);      // 1M x 128 f32 (512 MB)
  float*    acc_sum = (float*)(ws + 1024000000ULL);     // 40000 x 256 f32
  unsigned* acc_max = (unsigned*)(ws + 1064960000ULL);  // 40000 x 256 keys
  __bf16*   fr      = (__bf16*)(ws + 1105920000ULL);    // weight fragments
  __bf16* f_gc1s = fr;                                  // 5*3*8*512 = 61440
  __bf16* f_gc1r = f_gc1s + 61440;                      // 4*3*8*512 = 49152
  __bf16* f_gc2s = f_gc1r + 49152;                      // 5*4*8*512 = 81920
  __bf16* f_gc2r = f_gc2s + 81920;                      // 4*4*8*512 = 65536
  __bf16* f_dns  = f_gc2r + 65536;                      // 1*4*16*512 = 32768

  prep_frags<<<240, 256, 0, stream>>>(gc1_Wself, f_gc1s, 75, 128, 3, 5);
  prep_frags<<<192, 256, 0, stream>>>(gc1_Wrel,  f_gc1r, 75, 128, 3, 4);
  prep_frags<<<320, 256, 0, stream>>>(gc2_Wself, f_gc2s, 128, 128, 4, 5);
  prep_frags<<<256, 256, 0, stream>>>(gc2_Wrel,  f_gc2r, 128, 128, 4, 4);
  prep_frags<<<128, 256, 0, stream>>>(dense_W,   f_dns,  128, 256, 4, 1);
  init_accum<<<40000, 256, 0, stream>>>(acc_sum, acc_max);

  gc_kernel<75, 96><<<NBLK, 256, 0, stream>>>(atom_features, adj1, adj2, adj3, adj4,
      f_gc1s, f_gc1r, gc1_b, bn1_gamma, bn1_beta, bn1_mean, bn1_var, bufA);
  pool_kernel<<<125000, 256, 0, stream>>>(bufA, adj1, adj2, adj3, adj4, bufB);
  gc_kernel<128, 128><<<NBLK, 256, 0, stream>>>(bufB, adj1, adj2, adj3, adj4,
      f_gc2s, f_gc2r, gc2_b, bn1_gamma, bn1_beta, bn1_mean, bn1_var, bufA);
  pool_kernel<<<125000, 256, 0, stream>>>(bufA, adj1, adj2, adj3, adj4, bufB);
  dense_gather_kernel<<<NBLK, 256, 0, stream>>>(bufB, membership, f_dns, dense_b,
      bn3_gamma, bn3_beta, bn3_mean, bn3_var, acc_sum, acc_max);
  finalize_kernel<<<80000, 256, 0, stream>>>(acc_sum, acc_max, (float*)d_out);
}
